// SelfAttention_7748121002590
// MI455X (gfx1250) — matrix operand order repaired and checked by reading, not compiled, not run on hardware
//
#include <hip/hip_runtime.h>
#include <hip/hip_bf16.h>

typedef __attribute__((ext_vector_type(16))) __bf16 v16bf;
typedef __attribute__((ext_vector_type(8)))  __bf16 v8bf;
typedef __attribute__((ext_vector_type(8)))  float  v8f;

#define B_ 32
#define T_ 2048
#define H_ 512
#define S_ 512
#define R_ 8
#define XS 520   // padded LDS row stride (bf16 elems) -> conflict-free A-frag b128 loads
#define K3S 40   // kernel-3 transposed LDS stride (bf16 elems) -> hits all 64 banks

static __device__ inline v16bf cat8(v8bf lo, v8bf hi) {
  return __builtin_shufflevector(lo, hi, 0,1,2,3,4,5,6,7,8,9,10,11,12,13,14,15);
}

// A-matrix fragment (16x32 bf16) from a row-major [16 x stride] source.
// Layout: row M = lane%16; lane<16: elems0-15=K0..15;
// lane>=16: elems0-15=K16..31 (matches B fragment K order).
static __device__ inline v16bf load_a_rowmajor(const __bf16* base, int stride,
                                               int k0, int lane) {
  int row  = lane & 15;
  int half = lane >> 4;
  const __bf16* p = base + (size_t)row * stride + k0 + half * 16;
  v8bf lo = *(const v8bf*)p;
  v8bf hi = *(const v8bf*)(p + 8);
  return cat8(lo, hi);
}

// B-matrix fragment (32x16 bf16): column N = lane%16; lane<16 holds K=k0..k0+15,
// lane>=16 holds K=k0+16..k0+31. Source is row-major [N x stride] so K is contiguous.
static __device__ inline v16bf load_b_colK(const __bf16* base, int stride,
                                           int n0, int k0, int lane) {
  int n    = lane & 15;
  int half = lane >> 4;
  const __bf16* p = base + (size_t)(n0 + n) * stride + k0 + half * 16;
  v8bf lo = *(const v8bf*)p;
  v8bf hi = *(const v8bf*)(p + 8);
  return cat8(lo, hi);
}

// ---------------- Kernel 0: convert W1/W2 to bf16 ----------------
__global__ __launch_bounds__(256) void convert_params(
    const float* __restrict__ W1, const float* __restrict__ W2,
    __bf16* __restrict__ w1b, __bf16* __restrict__ w2b)
{
  int idx = blockIdx.x * 256 + threadIdx.x;
  if (idx < S_ * H_) w1b[idx] = (__bf16)W1[idx];
  if (idx < R_ * S_) w2b[idx] = (__bf16)W2[idx];
}

// ---------------- Kernel 1: score = tanh(x @ W1^T), logits = score @ W2^T ----
// One block = one batch b, 64 t-rows. Each of the 4 waves owns a private 16-row
// t-tile and a private LDS quadrant -> NO block barriers at all. Each wave:
//   stage own x-tile (f32 -> bf16),
//   32 s-tiles x 16 K-steps of v_wmma_f32_16x16x32_bf16 (W1 streamed from L2,
//   shared in time across the 4 waves -> 4x less L2 traffic than 16-row blocks),
//   fused tanhf, bf16 score strip to LDS,
//   16 more WMMAs folding the strip with W2^T into logits[16t x 8r].
__global__ __launch_bounds__(128) void attn_score_logits(
    const float* __restrict__ x, const __bf16* __restrict__ w1b,
    const __bf16* __restrict__ w2b, float* __restrict__ logits)
{
  __shared__ __bf16 xtile[4][16 * XS];   // per-wave x[b, t0..t0+15, :] as bf16
  __shared__ __bf16 stile[4][16 * XS];   // per-wave tanh-score strip [16t x 512s]
  const int tid  = threadIdx.x;
  const int wave = tid >> 5, lane = tid & 31;
  const int b    = blockIdx.y;
  const int t0   = blockIdx.x * 64 + wave * 16;
  const int half = lane >> 4, n = lane & 15;

  __bf16* xt = xtile[wave];
  __bf16* st = stile[wave];

  // Wave-local staging: 16 rows x 512 cols f32 -> bf16 (64 float4 per lane).
  const float* xrow = x + ((size_t)b * T_ + t0) * H_;
  #pragma unroll
  for (int j = 0; j < 64; ++j) {
    int f4  = lane + j * 32;          // 0..2047 float4s
    int row = f4 >> 7;                // t-row 0..15
    int col = (f4 & 127) * 4;         // h col
    float4 v = *(const float4*)(xrow + (size_t)row * H_ + col);
    __bf16* d = xt + row * XS + col;
    d[0] = (__bf16)v.x; d[1] = (__bf16)v.y; d[2] = (__bf16)v.z; d[3] = (__bf16)v.w;
  }
  // Same-wave LDS ops are in-order; no barrier needed (private quadrant).

  for (int i = 0; i < 32; ++i) {
    int s0 = i * 16;
    v8f acc = {};
    #pragma unroll
    for (int kt = 0; kt < H_ / 32; ++kt) {
      v16bf a  = load_a_rowmajor(xt, XS, kt * 32, lane);
      v16bf bm = load_b_colK(w1b, H_, s0, kt * 32, lane);
      acc = __builtin_amdgcn_wmma_f32_16x16x32_bf16(false, a, false, bm,
                                                    (short)0, acc, false, false);
    }
    // tanh + write bf16 score strip. C layout: VGPR v -> M = v + 8*half, N = n.
    #pragma unroll
    for (int v = 0; v < 8; ++v) {
      int m = v + half * 8;
      st[m * XS + s0 + n] = (__bf16)tanhf(acc[v]);
    }
  }

  // logits = score_strip @ W2^T (N padded to 16 by duplicating columns 8..15)
  v8f acc = {};
  #pragma unroll
  for (int kt = 0; kt < S_ / 32; ++kt) {
    v16bf a = load_a_rowmajor(st, XS, kt * 32, lane);
    const __bf16* p = w2b + (size_t)(n & 7) * S_ + kt * 32 + half * 16;
    v16bf bm = cat8(*(const v8bf*)p, *(const v8bf*)(p + 8));
    acc = __builtin_amdgcn_wmma_f32_16x16x32_bf16(false, a, false, bm,
                                                  (short)0, acc, false, false);
  }
  if (n < R_) {
    #pragma unroll
    for (int v = 0; v < 8; ++v) {
      int m = v + half * 8;  // t index within tile
      logits[((size_t)b * R_ + n) * T_ + t0 + m] = acc[v];
    }
  }
}

// ---------------- Kernel 2: softmax over T per (b,r) row ----------------
__global__ __launch_bounds__(256) void attn_softmax(
    const float* __restrict__ logits, float* __restrict__ wout,
    __bf16* __restrict__ wbf)
{
  __shared__ float red[256];
  const int row = blockIdx.x;        // b*R + r
  const int tid = threadIdx.x;
  const float* src = logits + (size_t)row * T_;
  float v[8];
  float mx = -3.0e38f;
  #pragma unroll
  for (int j = 0; j < 8; ++j) { v[j] = src[tid + j * 256]; mx = fmaxf(mx, v[j]); }
  red[tid] = mx; __syncthreads();
  for (int s = 128; s > 0; s >>= 1) {
    if (tid < s) red[tid] = fmaxf(red[tid], red[tid + s]);
    __syncthreads();
  }
  mx = red[0]; __syncthreads();
  float sum = 0.f;
  #pragma unroll
  for (int j = 0; j < 8; ++j) { v[j] = __expf(v[j] - mx); sum += v[j]; }
  red[tid] = sum; __syncthreads();
  for (int s = 128; s > 0; s >>= 1) {
    if (tid < s) red[tid] += red[tid + s];
    __syncthreads();
  }
  float inv = 1.0f / red[0];
  #pragma unroll
  for (int j = 0; j < 8; ++j) {
    float w = v[j] * inv;
    wout[(size_t)row * T_ + tid + j * 256] = w;
    wbf [(size_t)row * T_ + tid + j * 256] = (__bf16)w;
  }
}

// ---------------- Kernel 3: embed = weights @ x ----------------
// One block = one batch b, one 64-col H chunk; 4 waves, one 16-col N tile each.
// Per K-step: stage x 32x64 f32 -> bf16 transposed in LDS, then one WMMA.
__global__ __launch_bounds__(128) void attn_embed(
    const float* __restrict__ x, const __bf16* __restrict__ wbf,
    float* __restrict__ flat)
{
  __shared__ __bf16 xT[64 * K3S];    // [h][k] bf16, stride 40
  const int tid  = threadIdx.x;
  const int b    = blockIdx.y;
  const int h0   = blockIdx.x * 64;
  const int wave = tid >> 5, lane = tid & 31;
  const int n = lane & 15, half = lane >> 4;
  const __bf16* arow = wbf + ((size_t)b * R_ + (n & 7)) * T_; // rows 8..15 dup, unused
  v8f acc = {};

  for (int kt = 0; kt < T_ / 32; ++kt) {
    const int k0 = kt * 32;
    __syncthreads();
    #pragma unroll
    for (int i = 0; i < 4; ++i) {
      int f4  = tid + i * 128;       // 0..511 float4s of the 32x64 tile
      int row = f4 >> 4;             // k 0..31
      int col = (f4 & 15) * 4;       // h 0..63
      float4 v = *(const float4*)(x + ((size_t)b * T_ + k0 + row) * H_ + h0 + col);
      xT[(col + 0) * K3S + row] = (__bf16)v.x;
      xT[(col + 1) * K3S + row] = (__bf16)v.y;
      xT[(col + 2) * K3S + row] = (__bf16)v.z;
      xT[(col + 3) * K3S + row] = (__bf16)v.w;
    }
    __syncthreads();
    // A frag: weights row r = (lane%16)&7
    const __bf16* ap = arow + k0 + half * 16;
    v16bf a = cat8(*(const v8bf*)ap, *(const v8bf*)(ap + 8));
    // B frag: LDS column n of this wave's N tile, K contiguous
    const __bf16* bp = xT + (size_t)(wave * 16 + n) * K3S + half * 16;
    v16bf bm = cat8(*(const v8bf*)bp, *(const v8bf*)(bp + 8));
    acc = __builtin_amdgcn_wmma_f32_16x16x32_bf16(false, a, false, bm,
                                                  (short)0, acc, false, false);
  }
  if (half == 0) {                    // D rows 0..7 = embed[b, r, :]
    #pragma unroll
    for (int v = 0; v < 8; ++v) {
      flat[((size_t)b * R_ + v) * H_ + h0 + wave * 16 + n] = acc[v];
    }
  }
}

extern "C" void kernel_launch(void* const* d_in, const int* in_sizes, int n_in,
                              void* d_out, int out_size, void* d_ws, size_t ws_size,
                              hipStream_t stream) {
  const float* x  = (const float*)d_in[0];   // [B,T,H]
  const float* W1 = (const float*)d_in[1];   // [S,H]
  const float* W2 = (const float*)d_in[2];   // [R,S]

  char* ws = (char*)d_ws;
  __bf16* w1b    = (__bf16*)ws;                                   // 512 KB
  __bf16* w2b    = (__bf16*)(ws + (512u << 10));                  //   8 KB
  float*  logits = (float*) (ws + (512u << 10) + (16u << 10));    //   2 MB
  __bf16* wbf    = (__bf16*)(ws + (512u << 10) + (16u << 10) + (2u << 20)); // 1 MB

  float* flat = (float*)d_out;                 // [B, R*H] = 131072 f32
  float* wout = flat + (size_t)B_ * R_ * H_;   // [B, R, T] = 524288 f32

  convert_params   <<<dim3((S_ * H_ + 255) / 256), 256, 0, stream>>>(W1, W2, w1b, w2b);
  attn_score_logits<<<dim3(T_ / 64, B_),           128, 0, stream>>>(x, w1b, w2b, logits);
  attn_softmax     <<<dim3(B_ * R_),               256, 0, stream>>>(logits, wout, wbf);
  attn_embed       <<<dim3(H_ / 64, B_),           128, 0, stream>>>(x, wbf, flat);
}
